// PromptSwinTransformerBlock_25340307046825
// MI455X (gfx1250) — matrix operand order repaired and checked by reading, not compiled, not run on hardware
//
#include <hip/hip_runtime.h>
#include <hip/hip_bf16.h>
#include <math.h>

#define DIMC   96
#define NHEAD  3
#define WSZ    7
#define SHIFT_ 3
#define PP     5
#define HH_    56
#define WW_    56
#define HEADD  32
#define NWIN_B 64          // 8x8 windows per image
#define TPTOK  320         // PP * NWIN_B
#define NTOK   54          // PP + 49
#define NPAD   64
#define HIDDENC 384
#define BATCH  64
#define SEQ    3456        // H*W + TP
#define NWINT  (BATCH*NWIN_B)     // 4096
#define MROWS  (NWINT*NTOK)       // 221184
#define MTILES (MROWS/16)         // 13824
#define SCALEQ 0.17677669529663687f
#define BPITCH 24                 // LDS pitch for staged B panels (48B rows, 16B aligned)

typedef __attribute__((ext_vector_type(16))) __bf16 v16bf;
typedef __attribute__((ext_vector_type(8)))  float  v8f;

struct Pack8 { unsigned v[8]; };

static __device__ inline unsigned short f2bf(float f) {
    unsigned u = __builtin_bit_cast(unsigned, f);
    u += 0x7FFFu + ((u >> 16) & 1u);          // round-to-nearest-even
    return (unsigned short)(u >> 16);
}

// ---- WMMA operand gathers (per ISA 7.12.2 layouts, wave32) ----------------
// A matrix 16x32 bf16, src row-major with leading dim ld (elements).
static __device__ inline v16bf load_a(const unsigned short* src, int ld) {
    int lane = threadIdx.x & 31;
    int m    = lane & 15;
    int kh   = (lane >> 4) * 8;               // lanes 16-31 hold K+8 block
    Pack8 r;
#pragma unroll
    for (int v = 0; v < 8; ++v) {
        int k = ((v < 4) ? 2 * v : 16 + 2 * (v - 4)) + kh;
        r.v[v] = *(const unsigned*)(src + m * ld + k);   // bf16 pair (k,k+1)
    }
    return __builtin_bit_cast(v16bf, r);
}
// B matrix 32x16 bf16: lane%16 = n column, elements hold k in same order as load_a.
static __device__ inline v16bf load_b(const unsigned short* src, int ld) {
    int lane = threadIdx.x & 31;
    int n    = lane & 15;
    int kh   = (lane >> 4) * 8;
    struct { unsigned short e[16]; } r;
#pragma unroll
    for (int s = 0; s < 16; ++s) {
        int k = ((s < 8) ? s : 16 + (s - 8)) + kh;
        r.e[s] = src[k * ld + n];
    }
    return __builtin_bit_cast(v16bf, r);
}
static __device__ inline v8f wmma_bf16(v16bf a, v16bf b, v8f c) {
    return __builtin_amdgcn_wmma_f32_16x16x32_bf16(false, a, false, b,
                                                   (short)0, c, false, false);
}
static __device__ inline float wave_sum(float v) {
#pragma unroll
    for (int o = 16; o > 0; o >>= 1) v += __shfl_xor(v, o, 32);
    return v;
}

// ---- CDNA5 async global->LDS copy (ASYNCcnt) ------------------------------
// Copies 16 bytes per lane: LDS[lds_byte_addr] = MEM[gaddr .. gaddr+15].
static __device__ inline void async_copy_b128(unsigned lds_byte_addr, const void* gaddr) {
    asm volatile("global_load_async_to_lds_b128 %0, %1, off"
                 :: "v"(lds_byte_addr), "v"(gaddr) : "memory");
}
static __device__ inline void wait_async0() {
    asm volatile("s_wait_asynccnt 0x0" ::: "memory");
}

// ---- K0: fp32 -> bf16 ------------------------------------------------------
__global__ void k_f2bf(const float* __restrict__ s, unsigned short* __restrict__ d, int n) {
    int i = blockIdx.x * 256 + threadIdx.x;
    if (i < n) d[i] = f2bf(s[i]);
}

// ---- K1: LN1 + shifted-window/prompt gather -> bf16 xw --------------------
__global__ void k_ln_gather(const float* __restrict__ x, const float* __restrict__ g,
                            const float* __restrict__ b, unsigned short* __restrict__ xw) {
    int t    = blockIdx.x * 4 + (threadIdx.x >> 5);   // flat window-row
    int lane = threadIdx.x & 31;
    int win  = t / NTOK, n = t % NTOK;
    int bb   = win / NWIN_B, w = win % NWIN_B;
    int src;
    if (n < PP) {
        src = w * PP + n;
    } else {
        int j = n - PP, r = j / WSZ, c = j % WSZ;
        int hp = (w >> 3) * WSZ + r, wp = (w & 7) * WSZ + c;
        src = TPTOK + ((hp + SHIFT_) % HH_) * WW_ + ((wp + SHIFT_) % WW_);
    }
    const float* row = x + ((size_t)bb * SEQ + src) * DIMC;
    float v0 = row[lane], v1 = row[lane + 32], v2 = row[lane + 64];
    float mean = wave_sum(v0 + v1 + v2) * (1.f / 96.f);
    float d0 = v0 - mean, d1 = v1 - mean, d2 = v2 - mean;
    float inv = rsqrtf(wave_sum(d0 * d0 + d1 * d1 + d2 * d2) * (1.f / 96.f) + 1e-5f);
    unsigned short* o = xw + (size_t)t * DIMC;
    o[lane]      = f2bf(d0 * inv * g[lane]      + b[lane]);
    o[lane + 32] = f2bf(d1 * inv * g[lane + 32] + b[lane + 32]);
    o[lane + 64] = f2bf(d2 * inv * g[lane + 64] + b[lane + 64]);
}

// ---- K2: QKV GEMM, 8 waves/block share async-staged B panel in LDS --------
__global__ void k_qkv(const unsigned short* __restrict__ xw, const unsigned short* __restrict__ wq,
                      const float* __restrict__ bias, unsigned short* __restrict__ Q,
                      unsigned short* __restrict__ Kt, unsigned short* __restrict__ V) {
    __shared__ unsigned short Sb[96 * BPITCH];        // 4.5 KB staged weight panel
    int bid = blockIdx.x;
    int mtB = (bid / 18) * 8, nt = bid % 18;
    int tid = threadIdx.x, lane = tid & 31, wv = tid >> 5;

    if (tid < 192) {                                  // 96 rows x 2 halves, 16B each
        int k = tid >> 1, hf = tid & 1;
        async_copy_b128((unsigned)(uintptr_t)&Sb[k * BPITCH + hf * 8],
                        wq + (size_t)k * 288 + nt * 16 + hf * 8);
    }
    wait_async0();
    __syncthreads();

    int mt = mtB + wv;
    const unsigned short* ab = xw + (size_t)mt * 16 * DIMC;
    v16bf a0 = load_a(ab, DIMC), a1 = load_a(ab + 32, DIMC), a2 = load_a(ab + 64, DIMC);
    v8f c = {};
    c = wmma_bf16(a0, load_b(Sb + 0 * 32 * BPITCH, BPITCH), c);
    c = wmma_bf16(a1, load_b(Sb + 1 * 32 * BPITCH, BPITCH), c);
    c = wmma_bf16(a2, load_b(Sb + 2 * 32 * BPITCH, BPITCH), c);

    int n  = nt * 16 + (lane & 15);
    int s  = n / 96, hh = (n % 96) / 32, d = n & 31;  // uniform per block except d
    float bv = bias[n];
#pragma unroll
    for (int r = 0; r < 8; ++r) {
        int m = mt * 16 + r + 8 * (lane >> 4);
        int win = m / NTOK, tok = m % NTOK;
        float val = c[r] + bv;
        if (s == 0) val *= SCALEQ;
        unsigned short bf = f2bf(val);
        if      (s == 0) Q [((win * NHEAD + hh) * NPAD + tok) * HEADD + d] = bf;
        else if (s == 1) Kt[((win * NHEAD + hh) * HEADD + d) * NPAD + tok] = bf;
        else             V [((win * NHEAD + hh) * NPAD + tok) * HEADD + d] = bf;
    }
}

// ---- K3: per-(window,head) attention: S=qk^T, bias+mask, softmax, P@V -----
static __device__ inline int zone_(int p) { return p < 49 ? 0 : (p < 53 ? 1 : 2); }

__global__ void k_attn(const unsigned short* __restrict__ Q, const unsigned short* __restrict__ Kt,
                       const unsigned short* __restrict__ V, const float* __restrict__ rpb,
                       unsigned short* __restrict__ ao) {
    __shared__ float S[NPAD * NPAD];            // 16 KB
    __shared__ unsigned short Pm[NPAD * NPAD];  // 8 KB
    int bid  = blockIdx.x;
    int win  = bid / NHEAD, h = bid % NHEAD;
    int wloc = win % NWIN_B;
    int lane = threadIdx.x & 31;
    const unsigned short* qb = Q  + (size_t)(win * NHEAD + h) * NPAD * HEADD;
    const unsigned short* kb = Kt + (size_t)(win * NHEAD + h) * HEADD * NPAD;
    const unsigned short* vb = V  + (size_t)(win * NHEAD + h) * NPAD * HEADD;

    v16bf bk[4];
#pragma unroll
    for (int ntb = 0; ntb < 4; ++ntb) bk[ntb] = load_b(kb + ntb * 16, NPAD);
#pragma unroll
    for (int mt = 0; mt < 4; ++mt) {
        v16bf a = load_a(qb + mt * 16 * HEADD, HEADD);
#pragma unroll
        for (int ntb = 0; ntb < 4; ++ntb) {
            v8f c = {};
            c = wmma_bf16(a, bk[ntb], c);
#pragma unroll
            for (int r = 0; r < 8; ++r)
                S[(mt * 16 + r + 8 * (lane >> 4)) * NPAD + ntb * 16 + (lane & 15)] = c[r];
        }
    }
    __syncthreads();

    // bias + mask + softmax; each lane owns rows 2*lane, 2*lane+1
#pragma unroll
    for (int rr = 0; rr < 2; ++rr) {
        int i = lane * 2 + rr;
        float mx = -1e30f;
        for (int j = 0; j < NPAD; ++j) {
            float sc = S[i * NPAD + j];
            if (i < NTOK && j < NTOK) {
                if (i >= PP && j >= PP) {
                    int a = i - PP, bj = j - PP;
                    int ra = a / WSZ, ca = a % WSZ, rb = bj / WSZ, cb = bj % WSZ;
                    sc += rpb[((ra - rb + 6) * 13 + (ca - cb + 6)) * NHEAD + h];
                    int ha = (wloc >> 3) * WSZ + ra, wa = (wloc & 7) * WSZ + ca;
                    int hb = (wloc >> 3) * WSZ + rb, wb = (wloc & 7) * WSZ + cb;
                    if (3 * zone_(ha) + zone_(wa) != 3 * zone_(hb) + zone_(wb)) sc += -100.f;
                }
            } else sc = -1e30f;
            S[i * NPAD + j] = sc;
            mx = fmaxf(mx, sc);
        }
        float sum = 0.f;
        for (int j = 0; j < NPAD; ++j) {
            float e = __expf(S[i * NPAD + j] - mx);
            S[i * NPAD + j] = e; sum += e;
        }
        float inv = 1.f / sum;
        for (int j = 0; j < NPAD; ++j)
            Pm[i * NPAD + j] = f2bf(S[i * NPAD + j] * inv);
    }
    __syncthreads();

    v16bf bv4[4];
#pragma unroll
    for (int ntb = 0; ntb < 2; ++ntb)
#pragma unroll
        for (int kt = 0; kt < 2; ++kt)
            bv4[ntb * 2 + kt] = load_b(vb + kt * 32 * HEADD + ntb * 16, HEADD);
#pragma unroll
    for (int mt = 0; mt < 4; ++mt) {
#pragma unroll
        for (int ntb = 0; ntb < 2; ++ntb) {
            v8f c = {};
#pragma unroll
            for (int kt = 0; kt < 2; ++kt)
                c = wmma_bf16(load_a(Pm + mt * 16 * NPAD + kt * 32, NPAD), bv4[ntb * 2 + kt], c);
#pragma unroll
            for (int r = 0; r < 8; ++r) {
                int m = mt * 16 + r + 8 * (lane >> 4);
                if (m < NTOK)
                    ao[((size_t)(win * NTOK + m)) * DIMC + h * HEADD + ntb * 16 + (lane & 15)] = f2bf(c[r]);
            }
        }
    }
}

// ---- K4: proj GEMM (8 waves/block, async-staged B panel) + residual scatter
__global__ void k_proj(const unsigned short* __restrict__ ao, const unsigned short* __restrict__ wp,
                       const float* __restrict__ pb, const float* __restrict__ x,
                       float* __restrict__ y) {
    __shared__ unsigned short Sb[96 * BPITCH];
    int bid = blockIdx.x;
    int mtB = (bid / 6) * 8, nt = bid % 6;
    int tid = threadIdx.x, lane = tid & 31, wv = tid >> 5;

    if (tid < 192) {
        int k = tid >> 1, hf = tid & 1;
        async_copy_b128((unsigned)(uintptr_t)&Sb[k * BPITCH + hf * 8],
                        wp + (size_t)k * DIMC + nt * 16 + hf * 8);
    }
    wait_async0();
    __syncthreads();

    int mt = mtB + wv;
    const unsigned short* ab = ao + (size_t)mt * 16 * DIMC;
    v8f c = {};
    c = wmma_bf16(load_a(ab, DIMC),      load_b(Sb + 0 * 32 * BPITCH, BPITCH), c);
    c = wmma_bf16(load_a(ab + 32, DIMC), load_b(Sb + 1 * 32 * BPITCH, BPITCH), c);
    c = wmma_bf16(load_a(ab + 64, DIMC), load_b(Sb + 2 * 32 * BPITCH, BPITCH), c);

    int n = nt * 16 + (lane & 15);
    float bv = pb[n];
#pragma unroll
    for (int r = 0; r < 8; ++r) {
        int m = mt * 16 + r + 8 * (lane >> 4);
        int win = m / NTOK, tok = m % NTOK;
        int bb = win / NWIN_B, w = win % NWIN_B;
        int dst;
        if (tok < PP) {
            dst = w * PP + tok;
        } else {
            int j = tok - PP, r2 = j / WSZ, c2 = j % WSZ;
            int hp = (w >> 3) * WSZ + r2, wp2 = (w & 7) * WSZ + c2;
            dst = TPTOK + ((hp + SHIFT_) % HH_) * WW_ + ((wp2 + SHIFT_) % WW_);
        }
        size_t idx = ((size_t)bb * SEQ + dst) * DIMC + n;
        y[idx] = x[idx] + c[r] + bv;
    }
}

// ---- K5: LN2 -> bf16 -------------------------------------------------------
__global__ void k_ln_plain(const float* __restrict__ y, const float* __restrict__ g,
                           const float* __restrict__ b, unsigned short* __restrict__ xh) {
    int t = blockIdx.x * 4 + (threadIdx.x >> 5);
    int lane = threadIdx.x & 31;
    const float* row = y + (size_t)t * DIMC;
    float v0 = row[lane], v1 = row[lane + 32], v2 = row[lane + 64];
    float mean = wave_sum(v0 + v1 + v2) * (1.f / 96.f);
    float d0 = v0 - mean, d1 = v1 - mean, d2 = v2 - mean;
    float inv = rsqrtf(wave_sum(d0 * d0 + d1 * d1 + d2 * d2) * (1.f / 96.f) + 1e-5f);
    unsigned short* o = xh + (size_t)t * DIMC;
    o[lane]      = f2bf(d0 * inv * g[lane]      + b[lane]);
    o[lane + 32] = f2bf(d1 * inv * g[lane + 32] + b[lane + 32]);
    o[lane + 64] = f2bf(d2 * inv * g[lane + 64] + b[lane + 64]);
}

// ---- K6: fused MLP (fc1 + exact GELU in LDS + fc2) + residual -------------
__global__ void k_mlp(const unsigned short* __restrict__ xh, const unsigned short* __restrict__ w1,
                      const float* __restrict__ b1, const unsigned short* __restrict__ w2,
                      const float* __restrict__ b2, const float* __restrict__ y,
                      float* __restrict__ out) {
    __shared__ unsigned short Hs[2][16 * HIDDENC];   // 24 KB
    int wv = threadIdx.x >> 5;
    int lane = threadIdx.x & 31;
    int mt = blockIdx.x * 2 + wv;
    const unsigned short* ab = xh + (size_t)mt * 16 * DIMC;
    v16bf a0 = load_a(ab, DIMC), a1 = load_a(ab + 32, DIMC), a2 = load_a(ab + 64, DIMC);
#pragma unroll
    for (int nt = 0; nt < 24; ++nt) {
        v8f c = {};
        c = wmma_bf16(a0, load_b(w1 + (size_t)0  * HIDDENC + nt * 16, HIDDENC), c);
        c = wmma_bf16(a1, load_b(w1 + (size_t)32 * HIDDENC + nt * 16, HIDDENC), c);
        c = wmma_bf16(a2, load_b(w1 + (size_t)64 * HIDDENC + nt * 16, HIDDENC), c);
        float bb = b1[nt * 16 + (lane & 15)];
#pragma unroll
        for (int r = 0; r < 8; ++r) {
            float v = c[r] + bb;
            v = 0.5f * v * (1.f + erff(v * 0.70710678118654752f));   // exact GELU
            Hs[wv][(r + 8 * (lane >> 4)) * HIDDENC + nt * 16 + (lane & 15)] = f2bf(v);
        }
    }
    __syncthreads();
    v16bf af[12];
#pragma unroll
    for (int kt = 0; kt < 12; ++kt) af[kt] = load_a(&Hs[wv][kt * 32], HIDDENC);
#pragma unroll
    for (int nt = 0; nt < 6; ++nt) {
        v8f c = {};
#pragma unroll
        for (int kt = 0; kt < 12; ++kt)
            c = wmma_bf16(af[kt], load_b(w2 + (size_t)kt * 32 * DIMC + nt * 16, DIMC), c);
        int n = nt * 16 + (lane & 15);
        float bb = b2[n];
#pragma unroll
        for (int r = 0; r < 8; ++r) {
            size_t m = (size_t)mt * 16 + r + 8 * (lane >> 4);
            out[m * DIMC + n] = y[m * DIMC + n] + c[r] + bb;
        }
    }
}

// ---------------------------------------------------------------------------
extern "C" void kernel_launch(void* const* d_in, const int* in_sizes, int n_in,
                              void* d_out, int out_size, void* d_ws, size_t ws_size,
                              hipStream_t stream) {
    const float* x      = (const float*)d_in[0];
    const float* n1g    = (const float*)d_in[1];
    const float* n1b    = (const float*)d_in[2];
    const float* qkv_w  = (const float*)d_in[3];
    const float* qkv_b  = (const float*)d_in[4];
    const float* rpb    = (const float*)d_in[5];
    const float* proj_w = (const float*)d_in[6];
    const float* proj_b = (const float*)d_in[7];
    const float* n2g    = (const float*)d_in[8];
    const float* n2b    = (const float*)d_in[9];
    const float* fc1_w  = (const float*)d_in[10];
    const float* fc1_b  = (const float*)d_in[11];
    const float* fc2_w  = (const float*)d_in[12];
    const float* fc2_b  = (const float*)d_in[13];
    float* out = (float*)d_out;

    char* base = (char*)d_ws;
    size_t off = 0;
    auto alloc = [&](size_t bytes) {
        void* p = base + off;
        off = (off + bytes + 255) & ~(size_t)255;
        return p;
    };
    unsigned short* wqkv = (unsigned short*)alloc((size_t)96 * 288 * 2);
    unsigned short* wprj = (unsigned short*)alloc((size_t)96 * 96 * 2);
    unsigned short* wf1  = (unsigned short*)alloc((size_t)96 * 384 * 2);
    unsigned short* wf2  = (unsigned short*)alloc((size_t)384 * 96 * 2);
    unsigned short* xw   = (unsigned short*)alloc((size_t)MROWS * DIMC * 2);
    unsigned short* Q    = (unsigned short*)alloc((size_t)NWINT * NHEAD * NPAD * HEADD * 2);
    unsigned short* Kt   = (unsigned short*)alloc((size_t)NWINT * NHEAD * NPAD * HEADD * 2);
    unsigned short* V    = (unsigned short*)alloc((size_t)NWINT * NHEAD * NPAD * HEADD * 2);
    unsigned short* ao   = (unsigned short*)alloc((size_t)MROWS * DIMC * 2);
    float*          yb   = (float*)alloc((size_t)MROWS * DIMC * 4);
    unsigned short* xh   = (unsigned short*)alloc((size_t)MROWS * DIMC * 2);

    k_f2bf<<<(96 * 288 + 255) / 256, 256, 0, stream>>>(qkv_w, wqkv, 96 * 288);
    k_f2bf<<<(96 * 96 + 255) / 256, 256, 0, stream>>>(proj_w, wprj, 96 * 96);
    k_f2bf<<<(96 * 384 + 255) / 256, 256, 0, stream>>>(fc1_w, wf1, 96 * 384);
    k_f2bf<<<(384 * 96 + 255) / 256, 256, 0, stream>>>(fc2_w, wf2, 384 * 96);

    k_ln_gather<<<MROWS / 4, 128, 0, stream>>>(x, n1g, n1b, xw);
    k_qkv<<<(MTILES / 8) * 18, 256, 0, stream>>>(xw, wqkv, qkv_b, Q, Kt, V);
    k_attn<<<NWINT * NHEAD, 32, 0, stream>>>(Q, Kt, V, rpb, ao);
    k_proj<<<(MTILES / 8) * 6, 256, 0, stream>>>(ao, wprj, proj_b, x, yb);
    k_ln_plain<<<MROWS / 4, 128, 0, stream>>>(yb, n2g, n2b, xh);
    k_mlp<<<MTILES / 2, 64, 0, stream>>>(xh, wf1, fc1_b, wf2, fc2_b, yb, out);
}
